// CachedSelfAttention_66829691126401
// MI455X (gfx1250) — matrix operand order repaired and checked by reading, not compiled, not run on hardware
//
#include <hip/hip_runtime.h>

typedef __attribute__((ext_vector_type(16))) __bf16 v16bf;
typedef __attribute__((ext_vector_type(8)))  float  v8f;
typedef unsigned short ushort_t;

// Problem dims (fixed by the reference harness)
constexpr int BATCH = 4;
constexpr int SEQ   = 2048;
constexpr int DMODEL= 1024;
constexpr int NHEAD = 16;
constexpr int HDIM  = 64;       // head dim
constexpr int M_TOK = BATCH * SEQ;        // 8192 tokens
constexpr int N_QKV = 3 * DMODEL;         // 3072

// ---------------- helpers ----------------

__device__ __forceinline__ ushort_t f2bf(float f) {
    unsigned u = __builtin_bit_cast(unsigned, f);
    u += 0x7FFFu + ((u >> 16) & 1u);      // round-to-nearest-even
    return (ushort_t)(u >> 16);
}

__device__ __forceinline__ float bperm(float v, int srcLane) {
    return __builtin_bit_cast(float,
        __builtin_amdgcn_ds_bpermute(srcLane << 2, __builtin_bit_cast(int, v)));
}

__device__ __forceinline__ v8f wmma_bf16(v16bf a, v16bf b, v8f c) {
    return __builtin_amdgcn_wmma_f32_16x16x32_bf16(
        false, a, false, b, (short)0, c, false, false);
}

union Frag  { uint4 u[2]; v16bf v; };          // 32B operand fragment (8 VGPRs)
union PFrag { unsigned int u[8]; v16bf v; };   // packed-P fragment

// ---------------- fp32 -> bf16 convert ----------------

__global__ void f32_to_bf16_kernel(const float* __restrict__ in,
                                   ushort_t* __restrict__ out, int n) {
    int i = blockIdx.x * blockDim.x + threadIdx.x;
    int stride = gridDim.x * blockDim.x;
    for (; i < n; i += stride) out[i] = f2bf(in[i]);
}

// ---------------- QKV projection GEMM ----------------
// qkv[m,n] = sum_k xbf[m,k] * wbf[n,k] + bias[n]
// Writes Q,K as [B,H,S,hd] bf16, V transposed as [B,H,hd,S] bf16.
// Each wave computes a 32(M) x 64(N) block: B fragments reused across 2 A tiles.
__global__ __launch_bounds__(256)
void qkv_gemm_kernel(const ushort_t* __restrict__ xbf,
                     const ushort_t* __restrict__ wbf,
                     const float*    __restrict__ bias,
                     ushort_t* __restrict__ Qb,
                     ushort_t* __restrict__ Kb,
                     ushort_t* __restrict__ VTb) {
    const int wave = blockIdx.x * (blockDim.x >> 5) + (threadIdx.x >> 5);
    const int lane = threadIdx.x & 31;
    const int r    = lane & 15;
    const int hsel = lane >> 4;
    const int mtile = wave / (N_QKV / 64);     // 0..255 (32-row tiles)
    const int nblk  = wave % (N_QKV / 64);     // 0..47
    const int m0 = mtile * 32, n0 = nblk * 64;

    const ushort_t* arow0 = xbf + (size_t)(m0 + r) * DMODEL;
    const ushort_t* arow1 = xbf + (size_t)(m0 + 16 + r) * DMODEL;
    const ushort_t* wrow0 = wbf + (size_t)(n0 + r) * DMODEL;

    v8f c[2][4] = {};
    for (int kb = 0; kb < DMODEL; kb += 32) {
        Frag a0, a1;
        a0.u[0] = *(const uint4*)(arow0 + kb + 8 * hsel);
        a0.u[1] = *(const uint4*)(arow0 + kb + 16 + 8 * hsel);
        a1.u[0] = *(const uint4*)(arow1 + kb + 8 * hsel);
        a1.u[1] = *(const uint4*)(arow1 + kb + 16 + 8 * hsel);
#pragma unroll
        for (int i = 0; i < 4; ++i) {
            const ushort_t* wr = wrow0 + (size_t)i * 16 * DMODEL;
            Frag b;
            b.u[0] = *(const uint4*)(wr + kb + 8 * hsel);
            b.u[1] = *(const uint4*)(wr + kb + 16 + 8 * hsel);
            c[0][i] = wmma_bf16(a0.v, b.v, c[0][i]);
            c[1][i] = wmma_bf16(a1.v, b.v, c[1][i]);
        }
    }

#pragma unroll
    for (int ms = 0; ms < 2; ++ms) {
#pragma unroll
        for (int i = 0; i < 4; ++i) {
            const int n = n0 + i * 16 + r;     // output feature (fixed per lane)
            const int which = n >> 10;         // 0=Q 1=K 2=V
            const int f  = n & 1023;
            const int hh = f >> 6;
            const int d  = f & 63;
            const float bv = bias[n];
#pragma unroll
            for (int j = 0; j < 8; ++j) {
                const int m  = m0 + ms * 16 + j + 8 * hsel;  // token
                const int bb = m >> 11;
                const int s  = m & 2047;
                const ushort_t o = f2bf(c[ms][i][j] + bv);
                if (which == 0) {
                    Qb[((size_t)(bb * NHEAD + hh) * SEQ + s) * HDIM + d] = o;
                } else if (which == 1) {
                    Kb[((size_t)(bb * NHEAD + hh) * SEQ + s) * HDIM + d] = o;
                } else {
                    VTb[((size_t)(bb * NHEAD + hh) * HDIM + d) * SEQ + s] = o;
                }
            }
        }
    }
}

// ---------------- Flash attention ----------------
// One wave per (b,h,32-query tile). Computes S^T = K.Q^T so softmax rows are
// per-lane; P repacks in-lane into the A operand for P.V. V comes pre-transposed.
// K and V fragments are shared across the two 16-query sub-tiles.
__global__ __launch_bounds__(128)
void attn_kernel(const ushort_t* __restrict__ Qb,
                 const ushort_t* __restrict__ Kb,
                 const ushort_t* __restrict__ VTb,
                 ushort_t* __restrict__ Ob) {
    const int wave = blockIdx.x * (blockDim.x >> 5) + (threadIdx.x >> 5);
    const int lane = threadIdx.x & 31;
    const int r    = lane & 15;
    const int hsel = lane >> 4;
    const int qtile = wave & (SEQ / 32 - 1);   // 0..63 (32-query tiles)
    const int bh    = wave >> 6;               // 0..63
    const int q0 = qtile * 32;

    const ushort_t* qb  = Qb  + (size_t)bh * SEQ * HDIM;
    const ushort_t* kb  = Kb  + (size_t)bh * SEQ * HDIM;
    const ushort_t* vtb = VTb + (size_t)bh * HDIM * SEQ;

    // Resident B operands: Q rows, two 16-query sub-tiles x two 16x32 fragments.
    Frag bq[2][2];
#pragma unroll
    for (int qs = 0; qs < 2; ++qs)
#pragma unroll
        for (int p = 0; p < 2; ++p) {
            const ushort_t* qr =
                qb + (size_t)(q0 + qs * 16 + r) * HDIM + p * 32 + 8 * hsel;
            bq[qs][p].u[0] = *(const uint4*)(qr);
            bq[qs][p].u[1] = *(const uint4*)(qr + 16);
        }

    v8f co[2][4] = {};              // O accumulators (q in vgpr, d in lane)
    float m_run[2] = {-1e30f, -1e30f};
    float l_run[2] = {0.f, 0.f};

    for (int kc = 0; kc < SEQ; kc += 32) {
        // K fragments for this 32-key chunk (shared by both query sub-tiles)
        Frag ak[2][2];
#pragma unroll
        for (int st = 0; st < 2; ++st)
#pragma unroll
            for (int p = 0; p < 2; ++p) {
                const ushort_t* kr =
                    kb + (size_t)(kc + st * 16 + r) * HDIM + p * 32 + 8 * hsel;
                ak[st][p].u[0] = *(const uint4*)(kr);
                ak[st][p].u[1] = *(const uint4*)(kr + 16);
            }
        // prefetch next chunk of K
        __builtin_prefetch(kb + (size_t)(kc + 32 + r) * HDIM, 0, 1);

        // scores^T: per sub-tile, two 16(k) x 16(q) tiles, K-dim = 64
        v8f sc[2][2] = {};
#pragma unroll
        for (int qs = 0; qs < 2; ++qs)
#pragma unroll
            for (int st = 0; st < 2; ++st)
#pragma unroll
                for (int p = 0; p < 2; ++p)
                    sc[qs][st] = wmma_bf16(ak[st][p].v, bq[qs][p].v, sc[qs][st]);

        // V fragments for this chunk (shared by both query sub-tiles)
        Frag bv[4];
#pragma unroll
        for (int i = 0; i < 4; ++i) {
            const ushort_t* vr =
                vtb + (size_t)(i * 16 + r) * SEQ + kc + 8 * hsel;
            bv[i].u[0] = *(const uint4*)(vr);
            bv[i].u[1] = *(const uint4*)(vr + 16);
        }
        // prefetch next chunk of V
        __builtin_prefetch(vtb + (size_t)r * SEQ + kc + 32, 0, 1);

#pragma unroll
        for (int qs = 0; qs < 2; ++qs) {
            // online softmax; lane holds 16 of the 32 scores for
            // q = q0 + qs*16 + lane%16; partner lane (lane^16) holds the rest.
            float pf[16];
            float mc = -1e30f;
#pragma unroll
            for (int st = 0; st < 2; ++st)
#pragma unroll
                for (int j = 0; j < 8; ++j) {
                    float vsc = sc[qs][st][j] * 0.125f;   // hd^-0.5
                    pf[st * 8 + j] = vsc;
                    mc = fmaxf(mc, vsc);
                }
            mc = fmaxf(mc, bperm(mc, lane ^ 16));
            const float m_new = fmaxf(m_run[qs], mc);

            float sum = 0.f;
#pragma unroll
            for (int t = 0; t < 16; ++t) {
                float e = __expf(pf[t] - m_new);
                pf[t] = e;
                sum += e;
            }
            sum += bperm(sum, lane ^ 16);
            const float factor = __expf(m_run[qs] - m_new);
            l_run[qs] = l_run[qs] * factor + sum;
            m_run[qs] = m_new;

            // pack P into A-operand layout (all in-lane)
            PFrag ap;
#pragma unroll
            for (int j = 0; j < 4; ++j)
                ap.u[j] = (unsigned)f2bf(pf[2 * j]) |
                          ((unsigned)f2bf(pf[2 * j + 1]) << 16);
#pragma unroll
            for (int j = 0; j < 4; ++j)
                ap.u[4 + j] = (unsigned)f2bf(pf[8 + 2 * j]) |
                              ((unsigned)f2bf(pf[8 + 2 * j + 1]) << 16);

            // per-q rescale factors into O layout (q = j + 8*hsel)
            float fj[8];
#pragma unroll
            for (int j = 0; j < 8; ++j) fj[j] = bperm(factor, j + 8 * hsel);

#pragma unroll
            for (int i = 0; i < 4; ++i) {      // 4 d-tiles of 16
#pragma unroll
                for (int j = 0; j < 8; ++j) co[qs][i][j] *= fj[j];
                co[qs][i] = wmma_bf16(ap.v, bv[i].v, co[qs][i]);
            }
        }
    }

    const int bb = bh >> 4, hh = bh & 15;
#pragma unroll
    for (int qs = 0; qs < 2; ++qs) {
        const float inv = 1.0f / l_run[qs];
        float ij[8];
#pragma unroll
        for (int j = 0; j < 8; ++j) ij[j] = bperm(inv, j + 8 * hsel);
#pragma unroll
        for (int i = 0; i < 4; ++i)
#pragma unroll
            for (int j = 0; j < 8; ++j) {
                const int s = q0 + qs * 16 + j + 8 * hsel;
                Ob[((size_t)(bb * SEQ + s)) * DMODEL + hh * HDIM + i * 16 + r] =
                    f2bf(co[qs][i][j] * ij[j]);
            }
    }
}

// ---------------- Output projection GEMM ----------------
// out[m,n] = sum_k Ob[m,k] * owbf[n,k] + bias[n]   (fp32 output)
// 32(M) x 64(N) per wave.
__global__ __launch_bounds__(256)
void out_gemm_kernel(const ushort_t* __restrict__ obf,
                     const ushort_t* __restrict__ wbf,
                     const float*    __restrict__ bias,
                     float* __restrict__ out) {
    const int wave = blockIdx.x * (blockDim.x >> 5) + (threadIdx.x >> 5);
    const int lane = threadIdx.x & 31;
    const int r    = lane & 15;
    const int hsel = lane >> 4;
    const int mtile = wave / (DMODEL / 64);
    const int nblk  = wave % (DMODEL / 64);
    const int m0 = mtile * 32, n0 = nblk * 64;

    const ushort_t* arow0 = obf + (size_t)(m0 + r) * DMODEL;
    const ushort_t* arow1 = obf + (size_t)(m0 + 16 + r) * DMODEL;
    const ushort_t* wrow0 = wbf + (size_t)(n0 + r) * DMODEL;

    v8f c[2][4] = {};
    for (int kb = 0; kb < DMODEL; kb += 32) {
        Frag a0, a1;
        a0.u[0] = *(const uint4*)(arow0 + kb + 8 * hsel);
        a0.u[1] = *(const uint4*)(arow0 + kb + 16 + 8 * hsel);
        a1.u[0] = *(const uint4*)(arow1 + kb + 8 * hsel);
        a1.u[1] = *(const uint4*)(arow1 + kb + 16 + 8 * hsel);
#pragma unroll
        for (int i = 0; i < 4; ++i) {
            const ushort_t* wr = wrow0 + (size_t)i * 16 * DMODEL;
            Frag b;
            b.u[0] = *(const uint4*)(wr + kb + 8 * hsel);
            b.u[1] = *(const uint4*)(wr + kb + 16 + 8 * hsel);
            c[0][i] = wmma_bf16(a0.v, b.v, c[0][i]);
            c[1][i] = wmma_bf16(a1.v, b.v, c[1][i]);
        }
    }

#pragma unroll
    for (int ms = 0; ms < 2; ++ms)
#pragma unroll
        for (int i = 0; i < 4; ++i) {
            const int n = n0 + i * 16 + r;
            const float bv = bias[n];
#pragma unroll
            for (int j = 0; j < 8; ++j) {
                const int m = m0 + ms * 16 + j + 8 * hsel;
                out[(size_t)m * DMODEL + n] = c[ms][i][j] + bv;
            }
        }
}

// ---------------- host launcher ----------------

extern "C" void kernel_launch(void* const* d_in, const int* in_sizes, int n_in,
                              void* d_out, int out_size, void* d_ws, size_t ws_size,
                              hipStream_t stream) {
    const float* x     = (const float*)d_in[0];   // [4,2048,1024]
    const float* w_in  = (const float*)d_in[1];   // [3072,1024]
    const float* b_in  = (const float*)d_in[2];   // [3072]
    const float* w_out = (const float*)d_in[3];   // [1024,1024]
    const float* b_out = (const float*)d_in[4];   // [1024]
    float* out = (float*)d_out;

    // workspace carve-up (bf16 element counts)
    ushort_t* xbf  = (ushort_t*)d_ws;                 // 8,388,608
    ushort_t* wbf  = xbf  + (size_t)M_TOK * DMODEL;   // 3,145,728
    ushort_t* owbf = wbf  + (size_t)N_QKV * DMODEL;   // 1,048,576
    ushort_t* Qb   = owbf + (size_t)DMODEL * DMODEL;  // 8,388,608
    ushort_t* Kb   = Qb   + (size_t)M_TOK * DMODEL;
    ushort_t* VTb  = Kb   + (size_t)M_TOK * DMODEL;
    ushort_t* Ob   = VTb  + (size_t)M_TOK * DMODEL;   // total ~88 MB

    f32_to_bf16_kernel<<<4096, 256, 0, stream>>>(x,     xbf,  M_TOK * DMODEL);
    f32_to_bf16_kernel<<<2048, 256, 0, stream>>>(w_in,  wbf,  N_QKV * DMODEL);
    f32_to_bf16_kernel<<<1024, 256, 0, stream>>>(w_out, owbf, DMODEL * DMODEL);

    // (8192/32) * (3072/64) = 12288 waves, 8 waves per block
    qkv_gemm_kernel<<<12288 / 8, 256, 0, stream>>>(xbf, wbf, b_in, Qb, Kb, VTb);

    // 4*16*(2048/32) = 4096 waves, 4 waves per block
    attn_kernel<<<4096 / 4, 128, 0, stream>>>(Qb, Kb, VTb, Ob);

    // (8192/32) * (1024/64) = 4096 waves, 8 waves per block
    out_gemm_kernel<<<4096 / 8, 256, 0, stream>>>(Ob, owbf, b_out, out);
}
